// GNNEnergyNetwork_52226802319685
// MI455X (gfx1250) — matrix operand order repaired and checked by reading, not compiled, not run on hardware
//
#include <hip/hip_runtime.h>

typedef __bf16 bf16_t;
typedef __attribute__((ext_vector_type(16))) __bf16 v16bf;
typedef __attribute__((ext_vector_type(8)))  __bf16 v8bf;
typedef __attribute__((ext_vector_type(8)))  float  v8f;

#define WAVES   2      // waves per workgroup (independent)
#define BPW     2      // batch graphs per wave
#define NNODE   8
#define HDIM    256
#define NEDGE   56     // edges per graph (8*7)
#define ROWS    16     // BPW*NNODE rows per wave == one WMMA M-tile
#define EMT     7      // edge M-tiles per wave (112 rows)
#define LAYERS  3

// per-wave LDS block: h (bf16 16x256) | msg (bf16 112x256, aliased as f32 scratch) | agg (f32 16x256)
#define HB      (ROWS * HDIM * 2)          // 8192
#define MB      (EMT * 16 * HDIM * 2)      // 57344
#define AB      (ROWS * HDIM * 4)          // 16384
#define WBYTES  (HB + MB + AB)             // 81920

__device__ __forceinline__ void lds_fence() {
  // wave-internal LDS ordering: lanes exchange data through per-wave LDS slices
  asm volatile("s_wait_dscnt 0" ::: "memory");
}

// A-fragment (16x32 bf16) per CDNA5 layout: lane<16 -> M=lane, K {0..7,16..23};
// lane>=16 -> M=lane-16, K {8..15,24..31}. Two 16B chunks per lane.
__device__ __forceinline__ v16bf frag_from_bf16(const bf16_t* row, int col, int lane) {
  int kb = (lane & 16) ? 8 : 0;
  v8bf lo = *(const v8bf*)(row + col + kb);
  v8bf hi = *(const v8bf*)(row + col + kb + 16);
  return __builtin_shufflevector(lo, hi, 0,1,2,3,4,5,6,7,8,9,10,11,12,13,14,15);
}

__device__ __forceinline__ v16bf frag_from_f32(const float* row, int col, int lane) {
  int kb = (lane & 16) ? 8 : 0;
  v16bf r;
#pragma unroll
  for (int i = 0; i < 8; ++i) {
    r[i]     = (bf16_t)row[col + kb + i];
    r[i + 8] = (bf16_t)row[col + kb + 16 + i];
  }
  return r;
}

// pairs = [(i,j) for i in 0..7 for j in 0..7 if i!=j]; er in [0,112)
__device__ __forceinline__ void edge_rows(int er, int& rs, int& rd) {
  int bp = er / NEDGE;
  int e  = er - bp * NEDGE;
  int i  = e / 7;
  int j  = e - i * 7;
  j += (j >= i) ? 1 : 0;
  rs = bp * NNODE + i;
  rd = bp * NNODE + j;
}

// Reorder f32 weight [K,Ncols] into bf16 WMMA-B fragments:
// dst[((nt*KT+kt)*32+lane)*16+n] = W[kt*32+8*(lane>>4)+(n&7)+16*(n>>3)][nt*16+(lane&15)]
__global__ void swizzle_w(const float* __restrict__ W, bf16_t* __restrict__ dst,
                          int K, int Ncols) {
  int total = K * Ncols;
  int tid = blockIdx.x * blockDim.x + threadIdx.x;
  if (tid >= total) return;
  int KT   = K >> 5;
  int n    = tid & 15;
  int ln   = (tid >> 4) & 31;
  int rest = tid >> 9;          // nt*KT + kt
  int kt   = rest % KT;
  int nt   = rest / KT;
  dst[tid] = (bf16_t)W[(kt * 32 + ((ln >> 4) << 3) + (n & 7) + ((n >> 3) << 4)) * Ncols + nt * 16 + (ln & 15)];
}

__global__ __launch_bounds__(WAVES * 32)
void gnn_main(const float* __restrict__ z, const float* __restrict__ mod_emb,
              const float* __restrict__ init_b, const float* __restrict__ init_g, const float* __restrict__ init_be,
              const float* __restrict__ edge_b, const float* __restrict__ edge_g, const float* __restrict__ edge_be,
              const float* __restrict__ node_b, const float* __restrict__ node_g, const float* __restrict__ node_be,
              const float* __restrict__ un_b1, const float* __restrict__ un_W2, const float* __restrict__ un_b2,
              const float* __restrict__ bin_b1, const float* __restrict__ bin_W2, const float* __restrict__ bin_b2,
              const bf16_t* __restrict__ w_init, const bf16_t* __restrict__ w_edge,
              const bf16_t* __restrict__ w_node, const bf16_t* __restrict__ w_bin,
              const bf16_t* __restrict__ w_un,
              float* __restrict__ out) {
  extern __shared__ char smem[];
  const int lane = threadIdx.x & 31;
  const int wave = threadIdx.x >> 5;

  char* wb = smem + wave * WBYTES;
  bf16_t* hbuf = (bf16_t*)wb;
  bf16_t* msg  = (bf16_t*)(wb + HB);
  float*  scr  = (float*)(wb + HB);       // f32 alias of msg region (single-tile phases)
  float*  agg  = (float*)(wb + HB + MB);

  const int gb0  = (blockIdx.x * WAVES + wave) * BPW;  // first batch graph
  const int rsel = lane & 15;
  const int mofs = (lane & 16) >> 1;  // +8 for upper half-lanes (C/D layout)

  // per-lane edge row offsets (bf16 elements) for the 7 edge M-tiles
  int soff[EMT], doff[EMT];
#pragma unroll
  for (int mt = 0; mt < EMT; ++mt) {
    int rs, rd; edge_rows(mt * 16 + rsel, rs, rd);
    soff[mt] = rs * HDIM;
    doff[mt] = rd * HDIM;
  }

  // ---------------- init: h0 = relu(LN([z|emb] @ W_init + b)) -------------
  {
    bf16_t* X = (bf16_t*)agg;  // stage [16][96] bf16 in agg region
    for (int t = lane; t < ROWS * 96; t += 32) {
      int r = t / 96, c = t - r * 96;
      float v = (c < 64) ? z[((gb0 + (r >> 3)) * NNODE + (r & 7)) * 64 + c]
                         : mod_emb[(r & 7) * 32 + (c - 64)];
      X[r * 96 + c] = (bf16_t)v;
    }
    lds_fence();
    v16bf af[3];
#pragma unroll
    for (int kt = 0; kt < 3; ++kt) af[kt] = frag_from_bf16(X + rsel * 96, kt * 32, lane);
    const v16bf* W = (const v16bf*)w_init;
#pragma unroll
    for (int nt = 0; nt < 16; ++nt) {
      v8f acc = {};
#pragma unroll
      for (int kt = 0; kt < 3; ++kt)
        acc = __builtin_amdgcn_wmma_f32_16x16x32_bf16(false, af[kt], false,
                W[(nt * 3 + kt) * 32 + lane], (short)0, acc, false, false);
      int n = nt * 16 + rsel;
      float bv = init_b[n];
#pragma unroll
      for (int r = 0; r < 8; ++r) scr[(r + mofs) * HDIM + n] = acc[r] + bv;
    }
    lds_fence();
    {  // LN + relu -> hbuf (bf16)
      int rr = rsel, hh = (lane >> 4) * 128;
      float s = 0.f, q = 0.f;
      for (int c = 0; c < 128; c += 4) {
        float4 x = *(const float4*)&scr[rr * HDIM + hh + c];
        s += x.x + x.y + x.z + x.w;
        q += x.x * x.x + x.y * x.y + x.z * x.z + x.w * x.w;
      }
      s += __shfl_xor(s, 16, 32); q += __shfl_xor(q, 16, 32);
      float mean = s * (1.f / HDIM);
      float rstd = rsqrtf(q * (1.f / HDIM) - mean * mean + 1e-5f);
      for (int c = 0; c < 128; c += 4) {
        int col = hh + c;
        float4 x = *(const float4*)&scr[rr * HDIM + col];
        float4 g = *(const float4*)&init_g[col];
        float4 b = *(const float4*)&init_be[col];
        float y0 = (x.x - mean) * rstd * g.x + b.x; y0 = y0 > 0.f ? y0 : 0.f;
        float y1 = (x.y - mean) * rstd * g.y + b.y; y1 = y1 > 0.f ? y1 : 0.f;
        float y2 = (x.z - mean) * rstd * g.z + b.z; y2 = y2 > 0.f ? y2 : 0.f;
        float y3 = (x.w - mean) * rstd * g.w + b.w; y3 = y3 > 0.f ? y3 : 0.f;
        hbuf[rr * HDIM + col + 0] = (bf16_t)y0;
        hbuf[rr * HDIM + col + 1] = (bf16_t)y1;
        hbuf[rr * HDIM + col + 2] = (bf16_t)y2;
        hbuf[rr * HDIM + col + 3] = (bf16_t)y3;
      }
    }
    lds_fence();
  }

  // ---------------- message-passing layers --------------------------------
  for (int l = 0; l < LAYERS; ++l) {
    const v16bf* We = (const v16bf*)(w_edge + l * 512 * 256);
    const v16bf* Wn = (const v16bf*)(w_node + l * 512 * 256);
    const float* eb  = edge_b  + l * HDIM;
    const float* eg  = edge_g  + l * HDIM;
    const float* ebe = edge_be + l * HDIM;
    const float* nb  = node_b  + l * HDIM;
    const float* ng  = node_g  + l * HDIM;
    const float* nbe = node_be + l * HDIM;

    for (int t = lane; t < ROWS * HDIM; t += 32) agg[t] = 0.f;
    lds_fence();

    // ---- edge GEMM, flipped loops: each B-fragment reused across 7 M-tiles
    for (int nt = 0; nt < 16; ++nt) {
      v8f acc[EMT];
#pragma unroll
      for (int mt = 0; mt < EMT; ++mt) acc[mt] = (v8f){};
#pragma unroll
      for (int kt = 0; kt < 16; ++kt) {
        v16bf b = We[(nt * 16 + kt) * 32 + lane];
        int cb = (kt & 7) * 32;
#pragma unroll
        for (int mt = 0; mt < EMT; ++mt) {
          v16bf a = frag_from_bf16(hbuf + (kt < 8 ? soff[mt] : doff[mt]), cb, lane);
          acc[mt] = __builtin_amdgcn_wmma_f32_16x16x32_bf16(false, a, false, b,
                      (short)0, acc[mt], false, false);
        }
      }
      int n = nt * 16 + rsel;
      float bv = eb[n];
#pragma unroll
      for (int mt = 0; mt < EMT; ++mt)
#pragma unroll
        for (int r = 0; r < 8; ++r)
          msg[(mt * 16 + r + mofs) * HDIM + n] = (bf16_t)(acc[mt][r] + bv);
    }
    lds_fence();

    // ---- LN + relu + scatter-add into agg[dst] (per M-tile of msg)
    for (int mt = 0; mt < EMT; ++mt) {
      int rr = rsel, hh = (lane >> 4) * 128;
      const bf16_t* mrow = msg + (mt * 16 + rr) * HDIM;
      float s = 0.f, q = 0.f;
      for (int c = 0; c < 128; c += 8) {
        v8bf x8 = *(const v8bf*)(mrow + hh + c);
#pragma unroll
        for (int i = 0; i < 8; ++i) { float x = (float)x8[i]; s += x; q += x * x; }
      }
      s += __shfl_xor(s, 16, 32); q += __shfl_xor(q, 16, 32);
      float mean = s * (1.f / HDIM);
      float rstd = rsqrtf(q * (1.f / HDIM) - mean * mean + 1e-5f);
      int rs2, rd2; edge_rows(mt * 16 + rr, rs2, rd2);
      (void)rs2;
      float* arow = agg + rd2 * HDIM;
      for (int c = 0; c < 128; c += 8) {
        int col = hh + c;
        v8bf x8 = *(const v8bf*)(mrow + col);
        float4 g0 = *(const float4*)&eg[col],  g1 = *(const float4*)&eg[col + 4];
        float4 b0 = *(const float4*)&ebe[col], b1 = *(const float4*)&ebe[col + 4];
        float gg[8] = {g0.x, g0.y, g0.z, g0.w, g1.x, g1.y, g1.z, g1.w};
        float bb[8] = {b0.x, b0.y, b0.z, b0.w, b1.x, b1.y, b1.z, b1.w};
#pragma unroll
        for (int i = 0; i < 8; ++i) {
          float y = ((float)x8[i] - mean) * rstd * gg[i] + bb[i];
          y = y > 0.f ? y : 0.f;
          atomicAdd(arow + col + i, y);
        }
      }
    }
    lds_fence();

    // ---- node update: [h | agg] @ node_W -> LN -> relu -> + residual
    {
      v16bf af[16];
#pragma unroll
      for (int kt = 0; kt < 8; ++kt) af[kt]     = frag_from_bf16(hbuf + rsel * HDIM, kt * 32, lane);
#pragma unroll
      for (int kt = 0; kt < 8; ++kt) af[kt + 8] = frag_from_f32(agg + rsel * HDIM, kt * 32, lane);
#pragma unroll
      for (int nt = 0; nt < 16; ++nt) {
        v8f acc = {};
#pragma unroll
        for (int kt = 0; kt < 16; ++kt)
          acc = __builtin_amdgcn_wmma_f32_16x16x32_bf16(false, af[kt], false,
                  Wn[(nt * 16 + kt) * 32 + lane], (short)0, acc, false, false);
        int n = nt * 16 + rsel;
        float bv = nb[n];
#pragma unroll
        for (int r = 0; r < 8; ++r) scr[(r + mofs) * HDIM + n] = acc[r] + bv;
      }
      lds_fence();
      {
        int rr = rsel, hh = (lane >> 4) * 128;
        float s = 0.f, q = 0.f;
        for (int c = 0; c < 128; c += 4) {
          float4 x = *(const float4*)&scr[rr * HDIM + hh + c];
          s += x.x + x.y + x.z + x.w;
          q += x.x * x.x + x.y * x.y + x.z * x.z + x.w * x.w;
        }
        s += __shfl_xor(s, 16, 32); q += __shfl_xor(q, 16, 32);
        float mean = s * (1.f / HDIM);
        float rstd = rsqrtf(q * (1.f / HDIM) - mean * mean + 1e-5f);
        for (int c = 0; c < 128; ++c) {
          int col = hh + c;
          float x = scr[rr * HDIM + col];
          float y = (x - mean) * rstd * ng[col] + nbe[col];
          y = y > 0.f ? y : 0.f;
          float hn = y + (float)hbuf[rr * HDIM + col];
          hbuf[rr * HDIM + col] = (bf16_t)hn;
        }
      }
      lds_fence();
    }
  }

  // ---------------- unary head --------------------------------------------
  float usum = 0.f;
  {
    v16bf af[8];
#pragma unroll
    for (int kt = 0; kt < 8; ++kt) af[kt] = frag_from_bf16(hbuf + rsel * HDIM, kt * 32, lane);
    const v16bf* W = (const v16bf*)w_un;
#pragma unroll
    for (int nt = 0; nt < 2; ++nt) {
      v8f acc = {};
#pragma unroll
      for (int kt = 0; kt < 8; ++kt)
        acc = __builtin_amdgcn_wmma_f32_16x16x32_bf16(false, af[kt], false,
                W[(nt * 8 + kt) * 32 + lane], (short)0, acc, false, false);
      int n = nt * 16 + rsel;
      float bv = un_b1[n];
#pragma unroll
      for (int r = 0; r < 8; ++r) scr[(r + mofs) * 32 + n] = acc[r] + bv;
    }
    lds_fence();
    float s = 0.f;
    if (lane < 16) {
      for (int c = 0; c < 32; ++c) {
        float x = scr[lane * 32 + c];
        x = x > 0.f ? x : 0.f;
        s += x * un_W2[c];
      }
      s += un_b2[0];
    }
#pragma unroll
    for (int off = 1; off < 8; off <<= 1) s += __shfl_xor(s, off, 32);
    usum = s;  // lane0: batch0 unary; lane8: batch1 unary
    lds_fence();
  }

  // ---------------- binary head (flipped: B reused across 7 M-tiles) ------
  float bacc0 = 0.f, bacc1 = 0.f;
  {
    const v16bf* W = (const v16bf*)w_bin;
    float* scrB = scr;  // f32 [112][64] in msg region
    for (int nt = 0; nt < 4; ++nt) {
      v8f acc[EMT];
#pragma unroll
      for (int mt = 0; mt < EMT; ++mt) acc[mt] = (v8f){};
#pragma unroll
      for (int kt = 0; kt < 16; ++kt) {
        v16bf b = W[(nt * 16 + kt) * 32 + lane];
        int cb = (kt & 7) * 32;
#pragma unroll
        for (int mt = 0; mt < EMT; ++mt) {
          v16bf a = frag_from_bf16(hbuf + (kt < 8 ? soff[mt] : doff[mt]), cb, lane);
          acc[mt] = __builtin_amdgcn_wmma_f32_16x16x32_bf16(false, a, false, b,
                      (short)0, acc[mt], false, false);
        }
      }
      int n = nt * 16 + rsel;
      float bv = bin_b1[n];
#pragma unroll
      for (int mt = 0; mt < EMT; ++mt)
#pragma unroll
        for (int r = 0; r < 8; ++r)
          scrB[(mt * 16 + r + mofs) * 64 + n] = acc[mt][r] + bv;
    }
    lds_fence();
    for (int mt = 0; mt < EMT; ++mt) {
      int rr = rsel, hh = (lane >> 4) * 32;
      float s = 0.f;
      for (int c = 0; c < 32; ++c) {
        float x = scrB[(mt * 16 + rr) * 64 + hh + c];
        x = x > 0.f ? x : 0.f;
        s += x * bin_W2[hh + c];
      }
      s += __shfl_xor(s, 16, 32);
      if (lane < 16) {
        s += bin_b2[0];
        int bp = (mt * 16 + rr) / NEDGE;
        if (bp == 0) bacc0 += s; else bacc1 += s;
      }
    }
  }
#pragma unroll
  for (int off = 1; off < 16; off <<= 1) {
    bacc0 += __shfl_xor(bacc0, off, 32);
    bacc1 += __shfl_xor(bacc1, off, 32);
  }
  float u0 = __shfl(usum, 0, 32);
  float u1 = __shfl(usum, 8, 32);
  if (lane == 0) {
    out[gb0]     = u0 + bacc0;
    out[gb0 + 1] = u1 + bacc1;
  }
}

extern "C" void kernel_launch(void* const* d_in, const int* in_sizes, int n_in,
                              void* d_out, int out_size, void* d_ws, size_t ws_size,
                              hipStream_t stream) {
  const float* z       = (const float*)d_in[0];
  const float* mod_emb = (const float*)d_in[1];
  const float* init_W  = (const float*)d_in[2];
  const float* init_b  = (const float*)d_in[3];
  const float* init_g  = (const float*)d_in[4];
  const float* init_be = (const float*)d_in[5];
  const float* edge_W  = (const float*)d_in[6];
  const float* edge_b  = (const float*)d_in[7];
  const float* edge_g  = (const float*)d_in[8];
  const float* edge_be = (const float*)d_in[9];
  const float* node_W  = (const float*)d_in[10];
  const float* node_b  = (const float*)d_in[11];
  const float* node_g  = (const float*)d_in[12];
  const float* node_be = (const float*)d_in[13];
  const float* un_W1   = (const float*)d_in[14];
  const float* un_b1   = (const float*)d_in[15];
  const float* un_W2   = (const float*)d_in[16];
  const float* un_b2   = (const float*)d_in[17];
  const float* bin_W1  = (const float*)d_in[18];
  const float* bin_b1  = (const float*)d_in[19];
  const float* bin_W2  = (const float*)d_in[20];
  const float* bin_b2  = (const float*)d_in[21];
  float* out = (float*)d_out;

  // bf16 swizzled weights in workspace (~1.7 MB total)
  bf16_t* ws     = (bf16_t*)d_ws;
  bf16_t* w_init = ws;                          // 96*256
  bf16_t* w_edge = w_init + 96 * 256;           // 3*512*256
  bf16_t* w_node = w_edge + 3 * 512 * 256;      // 3*512*256
  bf16_t* w_bin  = w_node + 3 * 512 * 256;      // 512*64
  bf16_t* w_un   = w_bin + 512 * 64;            // 256*32
  (void)ws_size; (void)in_sizes; (void)n_in; (void)out_size;

  auto swz = [&](const float* W, bf16_t* dst, int K, int N) {
    int total = K * N;
    swizzle_w<<<(total + 255) / 256, 256, 0, stream>>>(W, dst, K, N);
  };
  swz(init_W, w_init, 96, 256);
  for (int l = 0; l < 3; ++l) {
    swz(edge_W + l * 512 * 256, w_edge + l * 512 * 256, 512, 256);
    swz(node_W + l * 512 * 256, w_node + l * 512 * 256, 512, 256);
  }
  swz(bin_W1, w_bin, 512, 64);
  swz(un_W1, w_un, 256, 32);

  const int B = 8192;
  size_t lds = (size_t)WAVES * WBYTES;  // 160 KB per workgroup
  gnn_main<<<B / (WAVES * BPW), WAVES * 32, lds, stream>>>(
      z, mod_emb,
      init_b, init_g, init_be,
      edge_b, edge_g, edge_be,
      node_b, node_g, node_be,
      un_b1, un_W2, un_b2,
      bin_b1, bin_W2, bin_b2,
      w_init, w_edge, w_node, w_bin, w_un,
      out);
}
